// SSIMLoss_42932493091333
// MI455X (gfx1250) — compile-verified, matrix operand order checked by reading, not run on hardware
//
#include <hip/hip_runtime.h>
#include <hip/hip_bf16.h>

// ---------------------------------------------------------------------------
// SSIM loss for 64 frames of 512x512 f32, 11x11 Gaussian window (sigma=1.5).
// Separable conv expressed as WMMA matrix sandwich: Out = Av * In * Bg with
// banded Gaussian matrices, run on v_wmma_f32_16x16x32_f16 (f32 accumulate).
// ---------------------------------------------------------------------------

typedef __attribute__((ext_vector_type(16))) _Float16 v16h;
typedef __attribute__((ext_vector_type(8)))  float    v8f;

#define FRAMES 64
#define HW     512
#define OUTDIM 502          // 512 - 11 + 1
#define TILES1D 32          // ceil(502/16)
#define TILES_PER_FRAME (TILES1D * TILES1D)   // 1024
#define NTILES (FRAMES * TILES_PER_FRAME)     // 65536
#define WPB 4                                 // waves (tiles) per block
#define NBLOCKS (NTILES / WPB)                // 16384

__device__ __constant__ float GW[11] = {
    0.0010284f, 0.0075988f, 0.0360008f, 0.1093607f, 0.2130055f,
    0.2660117f,
    0.2130055f, 0.1093607f, 0.0360008f, 0.0075988f, 0.0010284f
};

union H4U { _Float16 h[4]; uint2 u; };
union H8U { _Float16 h[8]; uint4 u; };
union V16U { v16h v; uint4 u[2]; };

__global__ void __launch_bounds__(128)
ssim_tile_kernel(const float* __restrict__ pred,
                 const float* __restrict__ targ,
                 float* __restrict__ partial)
{
    // per-wave private LDS regions (row pitch 40 ushorts = 80B, 16B aligned)
    __shared__ __align__(16) unsigned short sh_t[WPB][32][40];
    __shared__ __align__(16) unsigned short sh_p[WPB][32][40];
    __shared__ __align__(16) unsigned short sh_T[WPB][5][16][40];
    __shared__ float wsum[WPB];

    const int tid   = threadIdx.x;
    const int w     = tid >> 5;
    const int lane  = tid & 31;
    const int tile  = blockIdx.x * WPB + w;
    const int frame = tile >> 10;
    const int tin   = tile & 1023;
    const int oy0   = (tin >> 5) << 4;
    const int ox0   = (tin & 31) << 4;

    // ---------------- load 32x32 halo tile, convert f32 -> f16 -------------
    {
        const int gy = (oy0 + lane < 511) ? (oy0 + lane) : 511;
        const size_t rowbase = (size_t)frame * (HW * HW) + (size_t)gy * HW;
        #pragma unroll
        for (int c = 0; c < 8; ++c) {
            int col = ox0 + c * 4;
            col = (col < 508) ? col : 508;            // stay inside the row
            const float4 tv = *(const float4*)(targ + rowbase + col);
            const float4 pv = *(const float4*)(pred + rowbase + col);
            H4U ht, hp;
            ht.h[0] = (_Float16)tv.x; ht.h[1] = (_Float16)tv.y;
            ht.h[2] = (_Float16)tv.z; ht.h[3] = (_Float16)tv.w;
            hp.h[0] = (_Float16)pv.x; hp.h[1] = (_Float16)pv.y;
            hp.h[2] = (_Float16)pv.z; hp.h[3] = (_Float16)pv.w;
            *(uint2*)&sh_t[w][lane][c * 4] = ht.u;
            *(uint2*)&sh_p[w][lane][c * 4] = hp.u;
        }
    }

    // ---------------- banded Gaussian B-matrix fragment (32x16 f16) --------
    // Bg[k][n] = g[k-n] for 0 <= k-n <= 10 else 0.  B-fragment layout
    // mirrors the 16-bit A layout (ISA 7.12.2): lane -> N, VGPR v -> K pair.
    v16h bg;
    const int m   = lane & 15;                 // N (and A-matrix M)
    const int grp = (lane < 16) ? 0 : 8;
    #pragma unroll
    for (int v = 0; v < 8; ++v) {
        const int k0 = ((v & 3) << 1) + ((v < 4) ? 0 : 16) + grp;
        #pragma unroll
        for (int s = 0; s < 2; ++s) {
            const int d = (k0 + s) - m;
            float val = (d >= 0 && d <= 10) ? GW[d] : 0.0f;
            bg[v * 2 + s] = (_Float16)val;
        }
    }

    __syncthreads();

    // ---------------- stage-1 A fragments from LDS (16-bit A layout) -------
    const int off = grp;                       // ushort offset: 0 or 8
    auto loadA = [&](const unsigned short (*rows)[40], int rbase) -> v16h {
        V16U a;
        a.u[0] = *(const uint4*)&rows[rbase + m][off];
        a.u[1] = *(const uint4*)&rows[rbase + m][off + 16];
        return a.v;
    };

    const v16h at_t = loadA(sh_t[w], 0);       // targ rows 0-15
    const v16h at_b = loadA(sh_t[w], 16);      // targ rows 16-31
    const v16h ap_t = loadA(sh_p[w], 0);
    const v16h ap_b = loadA(sh_p[w], 16);

    // horizontal pass: T = In * Bg, stored transposed as sh_T[f][x][r]
    auto stage1 = [&](v16h a_top, v16h a_bot, int f) {
        v8f z0 = {}; v8f z1 = {};
        v8f c0 = __builtin_amdgcn_wmma_f32_16x16x32_f16(
                     false, a_top, false, bg, (short)0, z0, false, false);
        v8f c1 = __builtin_amdgcn_wmma_f32_16x16x32_f16(
                     false, a_bot, false, bg, (short)0, z1, false, false);
        H8U pk;
        #pragma unroll
        for (int i = 0; i < 8; ++i) pk.h[i] = (_Float16)c0[i];
        *(uint4*)&sh_T[w][f][m][off] = pk.u;           // rows off..off+7
        #pragma unroll
        for (int i = 0; i < 8; ++i) pk.h[i] = (_Float16)c1[i];
        *(uint4*)&sh_T[w][f][m][16 + off] = pk.u;      // rows 16+off..
    };

    stage1(at_t,        at_b,        0);   // targ        -> mu1
    stage1(ap_t,        ap_b,        1);   // pred        -> mu2
    stage1(at_t * at_t, at_b * at_b, 2);   // targ^2      (packed f16 mul)
    stage1(ap_t * ap_t, ap_b * ap_b, 3);   // pred^2
    stage1(at_t * ap_t, at_b * ap_b, 4);   // targ*pred

    __syncthreads();

    // ---------------- stage-2: Out^T = T^T * Bg (vertical pass) ------------
    v8f C[5];
    #pragma unroll
    for (int f = 0; f < 5; ++f) {
        V16U a;
        a.u[0] = *(const uint4*)&sh_T[w][f][m][off];
        a.u[1] = *(const uint4*)&sh_T[w][f][m][off + 16];
        v8f z = {};
        C[f] = __builtin_amdgcn_wmma_f32_16x16x32_f16(
                   false, a.v, false, bg, (short)0, z, false, false);
    }

    // ---------------- SSIM map + masked reduction --------------------------
    // C layout of Out^T: VGPR i, lane -> (x = i + grp + ox0, y = m + oy0)
    const float c1v = 0.0001f;   // (0.01)^2
    const float c2v = 0.0009f;   // (0.03)^2
    float lsum = 0.0f;
    if (oy0 + m < OUTDIM) {
        #pragma unroll
        for (int i = 0; i < 8; ++i) {
            const int ox = ox0 + i + grp;
            if (ox < OUTDIM) {
                const float mu1 = C[0][i], mu2 = C[1][i];
                const float et2 = C[2][i], ep2 = C[3][i], etp = C[4][i];
                const float mu1s = mu1 * mu1, mu2s = mu2 * mu2;
                const float m12  = mu1 * mu2;
                const float s1   = et2 - mu1s;
                const float s2   = ep2 - mu2s;
                const float s12  = etp - m12;
                const float num  = (2.0f * m12 + c1v) * (2.0f * s12 + c2v);
                const float den  = (mu1s + mu2s + c1v) * (s1 + s2 + c2v);
                lsum += num / den;
            }
        }
    }

    #pragma unroll
    for (int d = 16; d >= 1; d >>= 1)
        lsum += __shfl_xor(lsum, d, 32);
    if (lane == 0) wsum[w] = lsum;
    __syncthreads();
    if (tid == 0)
        partial[blockIdx.x] = wsum[0] + wsum[1] + wsum[2] + wsum[3];
}

__global__ void __launch_bounds__(256)
ssim_reduce_kernel(const float* __restrict__ partial, float* __restrict__ out,
                   int nb)
{
    __shared__ float s[256];
    float v = 0.0f;
    for (int i = threadIdx.x; i < nb; i += 256) v += partial[i];
    s[threadIdx.x] = v;
    __syncthreads();
    for (int st = 128; st > 0; st >>= 1) {
        if (threadIdx.x < st) s[threadIdx.x] += s[threadIdx.x + st];
        __syncthreads();
    }
    if (threadIdx.x == 0)
        out[0] = s[0] * (1.0f / ((float)FRAMES * (float)OUTDIM * (float)OUTDIM));
}

extern "C" void kernel_launch(void* const* d_in, const int* in_sizes, int n_in,
                              void* d_out, int out_size, void* d_ws, size_t ws_size,
                              hipStream_t stream)
{
    const float* pred = (const float*)d_in[0];   // "prediction"
    const float* targ = (const float*)d_in[1];   // "target"
    float* partial = (float*)d_ws;               // NBLOCKS floats = 64 KB

    ssim_tile_kernel<<<NBLOCKS, WPB * 32, 0, stream>>>(pred, targ, partial);
    ssim_reduce_kernel<<<1, 256, 0, stream>>>(partial, (float*)d_out, NBLOCKS);
}
